// CAWN_75067438399987
// MI455X (gfx1250) — compile-verified, matrix operand order checked by reading, not run on hardware
//
#include <hip/hip_runtime.h>
#include <hip/hip_bf16.h>

#define BB 1024
#define NNN 64
#define FF 128
#define DMM 512
#define DKK 256

typedef __attribute__((ext_vector_type(16))) _Float16 v16h;
typedef __attribute__((ext_vector_type(8)))  _Float16 v8h;
typedef __attribute__((ext_vector_type(8)))  float    v8f;

// ---------------------------------------------------------------------------
// K1: wq_red[h][d] = sum_e wq[h*DK+e][d]*wm[e];  wk_red[h][d] uses wm[DK+e]
// grid=4 (wq h0, wq h1, wk h0, wk h1), block=256
// ---------------------------------------------------------------------------
__global__ void k_reduce_w(const float* __restrict__ wq, const float* __restrict__ wk,
                           const float* __restrict__ wm,
                           float* __restrict__ wq_red, float* __restrict__ wk_red) {
  int blk = blockIdx.x;
  int h = blk & 1;
  const float* w = (blk < 2) ? wq : wk;
  const float* m = (blk < 2) ? wm : (wm + DKK);
  float* out = ((blk < 2) ? wq_red : wk_red) + h * DMM;
  int d0 = threadIdx.x;
  float a0 = 0.f, a1 = 0.f;
  for (int e = 0; e < DKK; ++e) {
    float mv = m[e];
    const float* row = w + (size_t)(h * DKK + e) * DMM;
    a0 += row[d0] * mv;
    a1 += row[d0 + 256] * mv;
  }
  out[d0] = a0;
  out[d0 + 256] = a1;
}

// ---------------------------------------------------------------------------
// K2: Wh[h][d'][k] = sum_e fc_w[d'][h*DK+e] * wv[h*DK+e][k], stored PRE-SWIZZLED
// into the per-lane WMMA B-fragment layout:
//   WhT[((h*32 + ntile)*16 + kstep)*32 + lane][16 halves]
// where lane = hi*16 + (d'&15), and within the 16 halves element 2*i+lb holds
// B[k][d'] with k = kstep*32 + (i&3)*2 + (i>>2)*16 + hi*8 + lb.
// grid = 2 heads * 32 d'-tiles = 64 blocks, block=256
// ---------------------------------------------------------------------------
__device__ __forceinline__ void store_b_frag(_Float16* __restrict__ WhT,
                                             int h, int nt, int k, int j, float val) {
  int kt = k >> 5;
  int k5 = k & 31;
  int lb = k5 & 1;
  int hi = (k5 >> 3) & 1;
  int i  = ((k5 >> 4) << 2) | ((k5 & 7) >> 1);
  int lane = hi * 16 + j;
  size_t idx = (((size_t)(h * 32 + nt) * 16 + kt) * 32 + lane) * 16 + 2 * i + lb;
  WhT[idx] = (_Float16)val;
}

__global__ void k_make_WhT(const float* __restrict__ fc_w, const float* __restrict__ wv,
                           _Float16* __restrict__ WhT) {
  int h  = blockIdx.x >> 5;
  int nt = blockIdx.x & 31;           // d' tile (16 wide)
  __shared__ float fcw[16][DKK];      // 16 KB
  int tid = threadIdx.x;
  for (int i = 0; i < 16; ++i) {
    int idx = tid + i * 256;          // 0..4095
    int j = idx >> 8, e = idx & 255;
    fcw[j][e] = fc_w[(size_t)(nt * 16 + j) * DMM + h * DKK + e];
  }
  __syncthreads();
  float acc[16][2];
  #pragma unroll
  for (int j = 0; j < 16; ++j) { acc[j][0] = 0.f; acc[j][1] = 0.f; }
  for (int e = 0; e < DKK; ++e) {
    const float* wvrow = wv + (size_t)(h * DKK + e) * DMM;
    float w0 = wvrow[tid], w1 = wvrow[tid + 256];
    #pragma unroll
    for (int j = 0; j < 16; ++j) {
      float f = fcw[j][e];
      acc[j][0] += f * w0;
      acc[j][1] += f * w1;
    }
  }
  for (int j = 0; j < 16; ++j) {
    store_b_frag(WhT, h, nt, tid,       j, acc[j][0]);
    store_b_frag(WhT, h, nt, tid + 256, j, acc[j][1]);
  }
}

// ---------------------------------------------------------------------------
// K3: transpose merge weights: m1_wt[e][d'] = m1_w[d'][e]; m2_wt likewise
// grid=384, block=256  (640*128 + 128*128 = 98304 elements)
// ---------------------------------------------------------------------------
__global__ void k_transpose_m(const float* __restrict__ m1_w, const float* __restrict__ m2_w,
                              float* __restrict__ m1_wt, float* __restrict__ m2_wt) {
  int i = blockIdx.x * 256 + threadIdx.x;
  if (i < 640 * 128) {
    int e = i >> 7, dp = i & 127;
    m1_wt[i] = m1_w[(size_t)dp * 640 + e];
  } else {
    int j = i - 640 * 128;
    int e = j >> 7, dp = j & 127;
    m2_wt[j] = m2_w[(size_t)dp * 128 + e];
  }
}

// ---------------------------------------------------------------------------
// K4: per-batch attention: scores -> masked softmax -> attn_out + kctx (f16)
// grid=1024 (one block per batch), block=512
// k[n][d] = concat(seq, seq_e, seq_t, seq_p)[d],  q[d] = concat(src,0,src_t,src_p)
// ---------------------------------------------------------------------------
__device__ __forceinline__ const float* comp_ptr(int comp, const float* c0, const float* c1,
                                                 const float* c2, const float* c3) {
  const float* p = c0;
  if (comp == 1) p = c1;
  else if (comp == 2) p = c2;
  else if (comp == 3) p = c3;
  return p;
}

__global__ void k_attn(const float* __restrict__ seq,   const float* __restrict__ seq_t,
                       const float* __restrict__ seq_e, const float* __restrict__ seq_p,
                       const float* __restrict__ src,   const float* __restrict__ src_t,
                       const float* __restrict__ src_p,
                       const unsigned char* __restrict__ mask,
                       const float* __restrict__ wq_red, const float* __restrict__ wk_red,
                       _Float16* __restrict__ Kctx, float* __restrict__ attn_out) {
  int b = blockIdx.x;
  int t = threadIdx.x;                       // 0..511
  __shared__ float p0[512], p1[512];
  __shared__ float sk0[NNN], sk1[NNN];
  __shared__ float sc[128];
  __shared__ float attn_s[2][NNN];
  __shared__ float wkr[2 * DMM];             // staged wk_red

  wkr[t] = wk_red[t];
  wkr[t + 512] = wk_red[t + 512];
  __syncthreads();

  // ---- pass A: score_k partials. thread -> (row n = t>>3, segment p = t&7)
  {
    int n = t >> 3, p = t & 7;
    int comp = p >> 1;                       // 0:seq 1:seq_e 2:seq_t 3:seq_p
    int off0 = (p & 1) * 64;
    const float* kp = comp_ptr(comp, seq, seq_e, seq_t, seq_p)
                      + ((size_t)b * NNN + n) * FF + off0;
    int dbase = comp * 128 + off0;
    float a0 = 0.f, a1 = 0.f;
    for (int j = 0; j < 64; ++j) {
      float v = kp[j];
      a0 += v * wkr[dbase + j];
      a1 += v * wkr[DMM + dbase + j];
    }
    p0[t] = a0; p1[t] = a1;
  }
  __syncthreads();
  if (t < NNN) {
    float s0 = 0.f, s1 = 0.f;
    for (int i = 0; i < 8; ++i) { s0 += p0[t * 8 + i]; s1 += p1[t * 8 + i]; }
    sk0[t] = s0; sk1[t] = s1;
  }
  __syncthreads();

  // ---- score_q = dot(q, wq_red[h]) via 512-wide tree reduction
  {
    int comp = t >> 7, o = t & 127;
    float qv;
    if (comp == 0) qv = src[(size_t)b * FF + o];
    else if (comp == 1) qv = 0.f;
    else if (comp == 2) qv = src_t[(size_t)b * FF + o];
    else qv = src_p[(size_t)b * FF + o];
    p0[t] = qv * wq_red[t];
    p1[t] = qv * wq_red[DMM + t];
  }
  __syncthreads();
  for (int s = 256; s > 0; s >>= 1) {
    if (t < s) { p0[t] += p0[t + s]; p1[t] += p1[t + s]; }
    __syncthreads();
  }
  float sq0 = p0[0], sq1 = p1[0];

  // ---- masked scores + per-head softmax (heads packed in 128 threads)
  if (t < 128) {
    int h = t >> 6, n = t & 63;
    float s = h ? (sk1[n] + sq1) : (sk0[n] + sq0);
    if (mask[(size_t)b * NNN + n]) s = -1e10f;
    sc[t] = s;
  }
  __syncthreads();
  if (t < 128) p0[t] = sc[t];
  __syncthreads();
  for (int s = 32; s > 0; s >>= 1) {                  // max
    if (t < 128 && (t & 63) < s) p0[t] = fmaxf(p0[t], p0[t + s]);
    __syncthreads();
  }
  if (t < 128) {
    float e = __expf(sc[t] - p0[(t >> 6) << 6]);
    sc[t] = e;
    p1[t] = e;
  }
  __syncthreads();
  for (int s = 32; s > 0; s >>= 1) {                  // sum
    if (t < 128 && (t & 63) < s) p1[t] += p1[t + s];
    __syncthreads();
  }
  if (t < 128) {
    int h = t >> 6, n = t & 63;
    float a = sc[t] / p1[h << 6];
    attn_s[h][n] = a;
    // attn_flat layout: [h*B + b, 1, NN]
    attn_out[((size_t)h * BB + b) * NNN + n] = a;
  }
  __syncthreads();

  // ---- pass B: kctx[h][d] = sum_n attn[h][n] * k[n][d]   (thread d = t)
  {
    int comp = t >> 7, o = t & 127;
    const float* kc = comp_ptr(comp, seq, seq_e, seq_t, seq_p) + (size_t)b * NNN * FF + o;
    float a0 = 0.f, a1 = 0.f;
    for (int n = 0; n < NNN; ++n) {
      float v = kc[(size_t)n * FF];
      a0 += attn_s[0][n] * v;
      a1 += attn_s[1][n] * v;
    }
    Kctx[(size_t)b * DMM + t]        = (_Float16)a0;        // head 0: [b][d]
    Kctx[(size_t)(BB + b) * DMM + t] = (_Float16)a1;        // head 1
  }
}

// ---------------------------------------------------------------------------
// K5: fc via WMMA:  y[b][d'] = LeakyReLU( sum_h Kctx_h[b] @ Wh_h^T + fc_b ) + q
// 2048 wave-tiles (64 M-tiles x 32 N-tiles), block=256 (8 waves), grid=256
// Inner loop: 2x b128 A loads + 2x b128 B loads (pre-swizzled) + 1 v_wmma.
// ---------------------------------------------------------------------------
__global__ void k_fc_wmma(const _Float16* __restrict__ Kctx, const _Float16* __restrict__ WhT,
                          const float* __restrict__ fc_b,
                          const float* __restrict__ src, const float* __restrict__ src_t,
                          const float* __restrict__ src_p,
                          float* __restrict__ y) {
  int wave = (blockIdx.x * blockDim.x + threadIdx.x) >> 5;
  int lane = threadIdx.x & 31;
  int tm = wave >> 5;                 // 0..63
  int tn = wave & 31;                 // 0..31
  int m0 = tm * 16, n0 = tn * 16;
  int lr = lane & 15;
  int hi = lane >> 4;

  v8f acc = {0.f, 0.f, 0.f, 0.f, 0.f, 0.f, 0.f, 0.f};
  #pragma unroll
  for (int h = 0; h < 2; ++h) {
    // A: row m0+lr of Kctx_h; per k-step the lane's halves are two contiguous
    // 8-half runs at k + hi*8 and k + 16 + hi*8.
    const _Float16* A = Kctx + (size_t)h * BB * DMM + (size_t)(m0 + lr) * DMM + (hi << 3);
    // B: pre-swizzled fragment stream: 16 contiguous halves per lane per k-step.
    const _Float16* Bsw = WhT + (((size_t)(h * 32 + tn) * 16) * 32 + lane) * 16;
    for (int kq = 0; kq < 16; ++kq) {       // K = 512, 32 per WMMA
      v8h alo = *(const v8h*)(A + kq * 32);
      v8h ahi = *(const v8h*)(A + kq * 32 + 16);
      v16h av = __builtin_shufflevector(alo, ahi, 0, 1, 2, 3, 4, 5, 6, 7,
                                        8, 9, 10, 11, 12, 13, 14, 15);
      v16h bv = *(const v16h*)(Bsw + (size_t)kq * 512);
      acc = __builtin_amdgcn_wmma_f32_16x16x32_f16(false, av, false, bv,
                                                   (short)0, acc, false, false);
    }
  }

  // epilogue: C/D layout: lane holds col=lr, rows m0 + j + 8*hi
  int col = n0 + lr;
  float bias = fc_b[col];
  int comp = col >> 7, off = col & 127;
  #pragma unroll
  for (int j = 0; j < 8; ++j) {
    int row = m0 + j + hi * 8;
    float v = acc[j] + bias;
    v = (v >= 0.f) ? v : 0.2f * v;            // LeakyReLU(0.2)
    float qv;
    if (comp == 0) qv = src[(size_t)row * FF + off];
    else if (comp == 1) qv = 0.f;
    else if (comp == 2) qv = src_t[(size_t)row * FF + off];
    else qv = src_p[(size_t)row * FF + off];
    y[(size_t)row * DMM + col] = v + qv;      // residual, pre-LayerNorm
  }
}

// ---------------------------------------------------------------------------
// K6: LayerNorm + MergeLayer.  grid=1024 (one row/batch), block=128
// ---------------------------------------------------------------------------
__global__ void k_merge(const float* __restrict__ y, const float* __restrict__ src,
                        const float* __restrict__ ln_g, const float* __restrict__ ln_b,
                        const float* __restrict__ m1_wt, const float* __restrict__ m1_b,
                        const float* __restrict__ m2_wt, const float* __restrict__ m2_b,
                        float* __restrict__ z) {
  int b = blockIdx.x, t = threadIdx.x;      // t in 0..127
  __shared__ float x[DMM + FF];             // 640
  __shared__ float red[128], red2[128];
  __shared__ float hdn[FF];

  float vals[4];
  float lsum = 0.f, lsq = 0.f;
  #pragma unroll
  for (int i = 0; i < 4; ++i) {
    float v = y[(size_t)b * DMM + t + i * 128];
    vals[i] = v; lsum += v; lsq += v * v;
  }
  red[t] = lsum; red2[t] = lsq;
  __syncthreads();
  for (int s = 64; s > 0; s >>= 1) {
    if (t < s) { red[t] += red[t + s]; red2[t] += red2[t + s]; }
    __syncthreads();
  }
  float mean = red[0] * (1.f / 512.f);
  float var  = red2[0] * (1.f / 512.f) - mean * mean;
  float inv  = rsqrtf(var + 1e-5f);
  #pragma unroll
  for (int i = 0; i < 4; ++i) {
    int d = t + i * 128;
    x[d] = (vals[i] - mean) * inv * ln_g[d] + ln_b[d];
  }
  x[DMM + t] = src[(size_t)b * FF + t];
  __syncthreads();

  float hacc = m1_b[t];
  for (int e = 0; e < DMM + FF; ++e) hacc += x[e] * m1_wt[(size_t)e * FF + t];
  hdn[t] = fmaxf(hacc, 0.f);
  __syncthreads();

  float zz = m2_b[t];
  for (int e = 0; e < FF; ++e) zz += hdn[e] * m2_wt[(size_t)e * FF + t];
  z[(size_t)b * FF + t] = zz;
}

// ---------------------------------------------------------------------------
extern "C" void kernel_launch(void* const* d_in, const int* in_sizes, int n_in,
                              void* d_out, int out_size, void* d_ws, size_t ws_size,
                              hipStream_t stream) {
  (void)in_sizes; (void)n_in; (void)out_size; (void)ws_size;
  const float* src   = (const float*)d_in[0];
  const float* src_t = (const float*)d_in[1];
  const float* src_p = (const float*)d_in[2];
  const float* seq   = (const float*)d_in[3];
  const float* seq_t = (const float*)d_in[4];
  const float* seq_e = (const float*)d_in[5];
  const float* seq_p = (const float*)d_in[6];
  const unsigned char* mask = (const unsigned char*)d_in[7];
  const float* wq   = (const float*)d_in[8];
  const float* wk   = (const float*)d_in[9];
  const float* wv   = (const float*)d_in[10];
  const float* wm   = (const float*)d_in[11];
  const float* fc_w = (const float*)d_in[12];
  const float* fc_b = (const float*)d_in[13];
  const float* ln_g = (const float*)d_in[14];
  const float* ln_b = (const float*)d_in[15];
  const float* m1_w = (const float*)d_in[16];
  const float* m1_b = (const float*)d_in[17];
  const float* m2_w = (const float*)d_in[18];
  const float* m2_b = (const float*)d_in[19];
  float* out = (float*)d_out;

  char* ws = (char*)d_ws;
  float*    wq_red = (float*)ws;      ws += 2 * DMM * sizeof(float);
  float*    wk_red = (float*)ws;      ws += 2 * DMM * sizeof(float);
  _Float16* WhT    = (_Float16*)ws;   ws += (size_t)2 * DMM * DMM * sizeof(_Float16);
  _Float16* Kctx   = (_Float16*)ws;   ws += (size_t)2 * BB * DMM * sizeof(_Float16);
  float*    y      = (float*)ws;      ws += (size_t)BB * DMM * sizeof(float);
  float*    m1_wt  = (float*)ws;      ws += (size_t)(DMM + FF) * FF * sizeof(float);
  float*    m2_wt  = (float*)ws;      ws += (size_t)FF * FF * sizeof(float);

  k_reduce_w<<<4, 256, 0, stream>>>(wq, wk, wm, wq_red, wk_red);
  k_make_WhT<<<64, 256, 0, stream>>>(fc_w, wv, WhT);
  k_transpose_m<<<384, 256, 0, stream>>>(m1_w, m2_w, m1_wt, m2_wt);
  k_attn<<<BB, 512, 0, stream>>>(seq, seq_t, seq_e, seq_p, src, src_t, src_p, mask,
                                 wq_red, wk_red, Kctx, out + (size_t)BB * FF);
  k_fc_wmma<<<256, 256, 0, stream>>>(Kctx, WhT, fc_b, src, src_t, src_p, y);
  k_merge<<<BB, 128, 0, stream>>>(y, src, ln_g, ln_b, m1_wt, m1_b, m2_wt, m2_b, out);
}
